// PointNetSetAbstraction_46420006535334
// MI455X (gfx1250) — matrix operand order repaired and checked by reading, not compiled, not run on hardware
//
#include <hip/hip_runtime.h>

// ---------------------------------------------------------------------------
// PointNet++ Set Abstraction for MI455X (gfx1250, wave32, WMMA)
// B=16, N=4096, S=1024, K=32, MLP 67->64->64->128, BN(train)+ReLU, max over K
// ---------------------------------------------------------------------------

#define B_   16
#define N_   4096
#define S_   1024
#define K_   32
#define M_   (B_ * S_ * K_)      // 524288 rows through the MLP
#define R2_  0.04f               // RADIUS^2
#define EPS_ 1e-5f
#define COORD_OUT_ (B_ * 3 * S_) // 49152 floats, first chunk of d_out

typedef __attribute__((ext_vector_type(16))) __bf16 v16bf;
typedef __attribute__((ext_vector_type(8)))  __bf16 v8bf;
typedef __attribute__((ext_vector_type(8)))  float  v8f;

__device__ __forceinline__ unsigned short f2bf(float f) {
    unsigned u = __float_as_uint(f);
    u += 0x7fffu + ((u >> 16) & 1u);     // round-to-nearest-even
    return (unsigned short)(u >> 16);
}

// ---------------------------------------------------------------------------
// 1) Farthest point sampling: one block per batch; xyz resident in LDS,
//    running min-distance in registers, block argmax via LDS tree.
//    Writes centers (B,S,3) and the new_coords output region (B,3,S).
// ---------------------------------------------------------------------------
__global__ void fps_kernel(const float* __restrict__ coords,
                           float* __restrict__ centers,
                           float* __restrict__ out) {
    __shared__ float sx[N_], sy[N_], sz[N_];
    __shared__ float rd[256];
    __shared__ int   ri[256];

    const int b = blockIdx.x;
    const int t = threadIdx.x;
    const float* cb = coords + (size_t)b * 3 * N_;
    for (int n = t; n < N_; n += 256) {
        sx[n] = cb[n];
        sy[n] = cb[N_ + n];
        sz[n] = cb[2 * N_ + n];
    }
    __syncthreads();

    float dist[16];
#pragma unroll
    for (int j = 0; j < 16; ++j) dist[j] = 3.4e38f;

    int cur = 0;
    for (int s = 0; s < S_; ++s) {
        const float cx = sx[cur], cy = sy[cur], cz = sz[cur];
        if (t == 0) {
            centers[((size_t)b * S_ + s) * 3 + 0] = cx;
            centers[((size_t)b * S_ + s) * 3 + 1] = cy;
            centers[((size_t)b * S_ + s) * 3 + 2] = cz;
            out[(size_t)b * 3 * S_ + 0 * S_ + s] = cx;
            out[(size_t)b * 3 * S_ + 1 * S_ + s] = cy;
            out[(size_t)b * 3 * S_ + 2 * S_ + s] = cz;
        }
        float bd = -1.0f; int bi = 0;
#pragma unroll
        for (int j = 0; j < 16; ++j) {
            const int n = j * 256 + t;
            const float dx = sx[n] - cx, dy = sy[n] - cy, dz = sz[n] - cz;
            const float d = dx * dx + dy * dy + dz * dz;
            dist[j] = fminf(dist[j], d);
            if (dist[j] > bd) { bd = dist[j]; bi = n; }
        }
        rd[t] = bd; ri[t] = bi;
        __syncthreads();
        for (int off = 128; off > 0; off >>= 1) {
            if (t < off) {
                const float od = rd[t + off]; const int oi = ri[t + off];
                if (od > rd[t] || (od == rd[t] && oi < ri[t])) { rd[t] = od; ri[t] = oi; }
            }
            __syncthreads();
        }
        cur = ri[0];
        __syncthreads();
    }
}

// ---------------------------------------------------------------------------
// 2) Ball query (kNN in-radius with nearest fallback): one thread per center,
//    32-entry sorted list fully unrolled (register resident), xyz tiled in LDS.
// ---------------------------------------------------------------------------
__global__ void ball_query_kernel(const float* __restrict__ coords,
                                  const float* __restrict__ centers,
                                  int* __restrict__ nbr) {
    __shared__ float px[1024], py[1024], pz[1024];
    const int t = threadIdx.x;                 // 64 threads
    const int cidx = blockIdx.x * 64 + t;      // center id in [0, B*S)
    const int b = cidx >> 10;                  // S_ == 1024
    const float cx = centers[(size_t)cidx * 3 + 0];
    const float cy = centers[(size_t)cidx * 3 + 1];
    const float cz = centers[(size_t)cidx * 3 + 2];

    float kd[K_]; int ki[K_];
#pragma unroll
    for (int j = 0; j < K_; ++j) { kd[j] = 3.4e38f; ki[j] = 0; }

    const float* cb = coords + (size_t)b * 3 * N_;
    for (int tile = 0; tile < N_ / 1024; ++tile) {
        for (int i = t; i < 1024; i += 64) {
            const int n = tile * 1024 + i;
            px[i] = cb[n]; py[i] = cb[N_ + n]; pz[i] = cb[2 * N_ + n];
        }
        __syncthreads();
        for (int p = 0; p < 1024; ++p) {
            const float dx = px[p] - cx, dy = py[p] - cy, dz = pz[p] - cz;
            const float d = dx * dx + dy * dy + dz * dz;
            if (d < kd[K_ - 1]) {
                float dc = d; int ic = tile * 1024 + p;
#pragma unroll
                for (int j = 0; j < K_; ++j) {
                    if (dc < kd[j]) {
                        const float td = kd[j]; const int ti = ki[j];
                        kd[j] = dc; ki[j] = ic; dc = td; ic = ti;
                    }
                }
            }
        }
        __syncthreads();
    }
    const int i0 = ki[0];
#pragma unroll
    for (int j = 0; j < K_; ++j)
        nbr[(size_t)cidx * K_ + j] = (kd[j] > R2_) ? i0 : ki[j];
}

// ---------------------------------------------------------------------------
// 3) Gather rows of the MLP input: [rel_xyz(3) | feat(64) | zeropad(29)] bf16,
//    row-major stride 96 (K padded 67 -> 96, multiple of 32 for WMMA).
// ---------------------------------------------------------------------------
__global__ void gather_kernel(const float* __restrict__ coords,
                              const float* __restrict__ feats,
                              const float* __restrict__ centers,
                              const int* __restrict__ nbr,
                              unsigned short* __restrict__ A) {
    const int row = blockIdx.x * blockDim.x + threadIdx.x;
    if (row >= M_) return;
    const int sc = row >> 5;                    // (b*S+s)
    const int b = sc >> 10;
    const int n = nbr[row];
    unsigned short* ar = A + (size_t)row * 96;
    const float* cb = coords + (size_t)b * 3 * N_;
    ar[0] = f2bf(cb[n]          - centers[(size_t)sc * 3 + 0]);
    ar[1] = f2bf(cb[N_ + n]     - centers[(size_t)sc * 3 + 1]);
    ar[2] = f2bf(cb[2 * N_ + n] - centers[(size_t)sc * 3 + 2]);
    const float* fb = feats + (size_t)b * 64 * N_ + n;
#pragma unroll 8
    for (int c = 0; c < 64; ++c) ar[3 + c] = f2bf(fb[(size_t)c * N_]);
#pragma unroll
    for (int c = 67; c < 96; ++c) ar[c] = 0;
}

// Pad/convert weights (COUT x CIN f32) -> (COUT x KPAD bf16, zero padded).
__global__ void wpad_kernel(const float* __restrict__ W,
                            unsigned short* __restrict__ Wp,
                            int cin, int kpad, int total) {
    const int i = blockIdx.x * blockDim.x + threadIdx.x;
    if (i >= total) return;
    const int o = i / kpad, k = i - o * kpad;
    Wp[i] = (k < cin) ? f2bf(W[(size_t)o * cin + k]) : (unsigned short)0;
}

// ---------------------------------------------------------------------------
// 4) GEMM: Y(M x COUT, f32) = A(M x KPAD, bf16) * W^T(KPAD x COUT, bf16)
//    via v_wmma_f32_16x16x32_bf16. 8 waves/block, each wave: 16 rows x COUT.
//    A per-lane layout (16-bit A 16x32): lane<16 -> K {kk..kk+7, kk+16..23},
//    lane>=16 -> K {kk+8..15, kk+24..31}. B per-lane: col N=lane%15..,
//    K same as A: {kk..kk+7, kk+16..23} (lanes 0-15) / {kk+8..15, kk+24..31} (lanes 16-31).
// ---------------------------------------------------------------------------
template <int KPAD, int COUT>
__global__ void gemm_bf16_wmma(const unsigned short* __restrict__ Ain,
                               const unsigned short* __restrict__ Win,
                               float* __restrict__ Y) {
    const __bf16* A = (const __bf16*)Ain;
    const __bf16* W = (const __bf16*)Win;
    const int lane = threadIdx.x & 31;
    const int wave = threadIdx.x >> 5;
    const int tileM = (blockIdx.x * 8 + wave) * 16;
    constexpr int NT = COUT / 16;

    v8f zero;
#pragma unroll
    for (int i = 0; i < 8; ++i) zero[i] = 0.0f;
    v8f acc[NT];
#pragma unroll
    for (int t = 0; t < NT; ++t) acc[t] = zero;

    const int halfsel = lane >> 4;
    const int l15 = lane & 15;
    const __bf16* aRow = A + (size_t)(tileM + l15) * KPAD;
    // speculative prefetch of the next block's A tile (global_prefetch_b8)
    __builtin_prefetch(aRow + (size_t)128 * KPAD, 0, 1);

#pragma unroll
    for (int kk = 0; kk < KPAD; kk += 32) {
        const int off0 = kk + halfsel * 8;
        const v8bf alo = *(const v8bf*)(aRow + off0);
        const v8bf ahi = *(const v8bf*)(aRow + off0 + 16);
        v16bf afrag;
#pragma unroll
        for (int i = 0; i < 8; ++i) { afrag[i] = alo[i]; afrag[i + 8] = ahi[i]; }
#pragma unroll
        for (int t = 0; t < NT; ++t) {
            const __bf16* wp = W + (size_t)(t * 16 + l15) * KPAD + off0;
            const v8bf blo = *(const v8bf*)(wp);
            const v8bf bhi = *(const v8bf*)(wp + 16);
            v16bf bfrag;
#pragma unroll
            for (int i = 0; i < 8; ++i) { bfrag[i] = blo[i]; bfrag[i + 8] = bhi[i]; }
            acc[t] = __builtin_amdgcn_wmma_f32_16x16x32_bf16(
                false, afrag, false, bfrag, (short)0, acc[t], false, false);
        }
    }
    // D layout: VGPR r -> row tileM + r + 8*halfsel, col = nt*16 + (lane&15)
    const int rbase = tileM + halfsel * 8;
#pragma unroll
    for (int t = 0; t < NT; ++t)
#pragma unroll
        for (int r = 0; r < 8; ++r)
            Y[(size_t)(rbase + r) * COUT + t * 16 + l15] = acc[t][r];
}

// ---------------------------------------------------------------------------
// 5) BatchNorm (training stats) — deterministic two-stage reduction.
//    Conv bias cancels under mean subtraction, so it is never read.
// ---------------------------------------------------------------------------
template <int C>
__global__ void bn_partial(const float* __restrict__ Y,
                           float* __restrict__ psum, float* __restrict__ psq,
                           int rows_per_block) {
    __shared__ float ss[256], ss2[256];
    constexpr int G = 256 / C;
    const int c = threadIdx.x % C;
    const int g = threadIdx.x / C;
    const int r0 = blockIdx.x * rows_per_block;
    float s = 0.0f, s2 = 0.0f;
    for (int r = r0 + g; r < r0 + rows_per_block; r += G) {
        const float v = Y[(size_t)r * C + c];
        s += v; s2 += v * v;
    }
    ss[threadIdx.x] = s; ss2[threadIdx.x] = s2;
    __syncthreads();
    if (g == 0) {
#pragma unroll
        for (int gg = 1; gg < G; ++gg) { s += ss[gg * C + c]; s2 += ss2[gg * C + c]; }
        psum[(size_t)blockIdx.x * C + c] = s;
        psq [(size_t)blockIdx.x * C + c] = s2;
    }
}

template <int C>
__global__ void bn_finalize(const float* __restrict__ psum,
                            const float* __restrict__ psq,
                            const float* __restrict__ gamma,
                            const float* __restrict__ beta,
                            float* __restrict__ ss, int parts, float invM) {
    const int c = threadIdx.x;
    float s = 0.0f, s2 = 0.0f;
    for (int p = 0; p < parts; ++p) { s += psum[(size_t)p * C + c]; s2 += psq[(size_t)p * C + c]; }
    const float mu = s * invM;
    const float var = s2 * invM - mu * mu;
    const float sc = gamma[c] * rsqrtf(var + EPS_);
    ss[c] = sc;
    ss[C + c] = beta[c] - mu * sc;
}

template <int C>
__global__ void bn_apply_bf16(const float* __restrict__ Y,
                              const float* __restrict__ ss,
                              unsigned short* __restrict__ X) {
    const int i = blockIdx.x * blockDim.x + threadIdx.x;
    const int c = i & (C - 1);
    const float v = fmaxf(Y[i] * ss[c] + ss[C + c], 0.0f);
    X[i] = f2bf(v);
}

// 6) Fused layer-3 affine+ReLU + max over K + output transpose (B,128,S).
__global__ void maxpool_out(const float* __restrict__ Y,
                            const float* __restrict__ ss,
                            float* __restrict__ out) {
    const int tid = blockIdx.x * blockDim.x + threadIdx.x; // B*S*128 threads
    const int c = tid & 127;
    const int s = (tid >> 7) & 1023;
    const int b = tid >> 17;
    const float sc = ss[c], sh = ss[128 + c];
    const size_t m0 = ((size_t)b * S_ + s) * K_;
    float mx = 0.0f;                                       // ReLU floor
#pragma unroll
    for (int k = 0; k < K_; ++k)
        mx = fmaxf(mx, Y[(m0 + k) * 128 + c] * sc + sh);
    out[COORD_OUT_ + ((size_t)b * 128 + c) * S_ + s] = mx;
}

// ---------------------------------------------------------------------------
extern "C" void kernel_launch(void* const* d_in, const int* in_sizes, int n_in,
                              void* d_out, int out_size, void* d_ws, size_t ws_size,
                              hipStream_t stream) {
    (void)in_sizes; (void)n_in; (void)out_size; (void)ws_size;
    const float* coords = (const float*)d_in[0];   // (B,3,N)
    const float* feats  = (const float*)d_in[1];   // (B,64,N)
    const float* W1 = (const float*)d_in[2];       // (64,67)
    const float* g1 = (const float*)d_in[4];
    const float* be1 = (const float*)d_in[5];
    const float* W2 = (const float*)d_in[6];       // (64,64)
    const float* g2 = (const float*)d_in[8];
    const float* be2 = (const float*)d_in[9];
    const float* W3 = (const float*)d_in[10];      // (128,64)
    const float* g3 = (const float*)d_in[12];
    const float* be3 = (const float*)d_in[13];
    float* out = (float*)d_out;

    char* p = (char*)d_ws;
    auto alloc = [&](size_t bytes) -> void* {
        void* r = (void*)p; p += (bytes + 255) & ~(size_t)255; return r;
    };
    unsigned short* bufA = (unsigned short*)alloc((size_t)M_ * 96 * 2); // activations bf16
    float* bufB          = (float*)alloc((size_t)M_ * 128 * 4);        // GEMM out f32
    float* centers       = (float*)alloc((size_t)B_ * S_ * 3 * 4);
    int*   nbr           = (int*)alloc((size_t)B_ * S_ * K_ * 4);
    unsigned short* Wp1  = (unsigned short*)alloc(64 * 96 * 2);
    unsigned short* Wp2  = (unsigned short*)alloc(64 * 64 * 2);
    unsigned short* Wp3  = (unsigned short*)alloc(128 * 64 * 2);
    float* psum          = (float*)alloc(512 * 128 * 4);
    float* psq           = (float*)alloc(512 * 128 * 4);
    float* ss            = (float*)alloc(2 * 128 * 4);

    const float invM = 1.0f / (float)M_;

    fps_kernel<<<B_, 256, 0, stream>>>(coords, centers, out);
    ball_query_kernel<<<(B_ * S_) / 64, 64, 0, stream>>>(coords, centers, nbr);
    gather_kernel<<<M_ / 256, 256, 0, stream>>>(coords, feats, centers, nbr, bufA);
    wpad_kernel<<<(64 * 96 + 255) / 256, 256, 0, stream>>>(W1, Wp1, 67, 96, 64 * 96);
    wpad_kernel<<<(64 * 64 + 255) / 256, 256, 0, stream>>>(W2, Wp2, 64, 64, 64 * 64);
    wpad_kernel<<<(128 * 64 + 255) / 256, 256, 0, stream>>>(W3, Wp3, 64, 64, 128 * 64);

    // Layer 1: 67(->96) -> 64
    gemm_bf16_wmma<96, 64><<<M_ / 128, 256, 0, stream>>>(bufA, Wp1, bufB);
    bn_partial<64><<<512, 256, 0, stream>>>(bufB, psum, psq, M_ / 512);
    bn_finalize<64><<<1, 64, 0, stream>>>(psum, psq, g1, be1, ss, 512, invM);
    bn_apply_bf16<64><<<(M_ * 64) / 256, 256, 0, stream>>>(bufB, ss, bufA);

    // Layer 2: 64 -> 64
    gemm_bf16_wmma<64, 64><<<M_ / 128, 256, 0, stream>>>(bufA, Wp2, bufB);
    bn_partial<64><<<512, 256, 0, stream>>>(bufB, psum, psq, M_ / 512);
    bn_finalize<64><<<1, 64, 0, stream>>>(psum, psq, g2, be2, ss, 512, invM);
    bn_apply_bf16<64><<<(M_ * 64) / 256, 256, 0, stream>>>(bufB, ss, bufA);

    // Layer 3: 64 -> 128, then fused BN+ReLU+max-over-K into output
    gemm_bf16_wmma<64, 128><<<M_ / 128, 256, 0, stream>>>(bufA, Wp3, bufB);
    bn_partial<128><<<512, 256, 0, stream>>>(bufB, psum, psq, M_ / 512);
    bn_finalize<128><<<1, 128, 0, stream>>>(psum, psq, g3, be3, ss, 512, invM);
    maxpool_out<<<(B_ * S_ * 128) / 256, 256, 0, stream>>>(bufB, ss, out);
}
